// SpatialAttentionBlock_34050500722775
// MI455X (gfx1250) — matrix operand order repaired and checked by reading, not compiled, not run on hardware
//
#include <hip/hip_runtime.h>
#include <hip/hip_bf16.h>
#include <math.h>

// Problem constants (match reference)
#define BB   8
#define TT   2048
#define CC   256
#define FDD  256
#define KK   128
#define VV   256
#define OUTC 512   // C + V

typedef __attribute__((ext_vector_type(16))) _Float16 v16h;
typedef __attribute__((ext_vector_type(8)))  float    v8f;

// ---------------------------------------------------------------------------
// WMMA fragment helpers (wave32; A/B in this code's contiguous k order, not the ISA table order)
// A 16x32 f16: lane L -> row m=L%16, half=L/16; VGPR r holds K = kbase,kbase+1
//   kbase = 16*half + 2r
// B 32x16 f16: lane L -> col n=L%16, kg=L/16; VGPR r holds K = kg*16+2r, +1
// C/D 16x16 f32: lane L -> col n=L%16; VGPR r holds row m = r + 8*(L/16)
// ---------------------------------------------------------------------------
__device__ __forceinline__ int a_kbase(int r, int h) {
    return 16 * h + 2 * r;
}

__device__ __forceinline__ v16h load_a_f32(const float* a0, int ld, int lane) {
    const int m = lane & 15, h = lane >> 4;
    const float* row = a0 + (size_t)m * ld;
    v16h a;
#pragma unroll
    for (int r = 0; r < 8; ++r) {
        const int kb = a_kbase(r, h);
        a[2 * r]     = (_Float16)row[kb];
        a[2 * r + 1] = (_Float16)row[kb + 1];
    }
    return a;
}

__device__ __forceinline__ v16h load_a_f16(const _Float16* a0, int ld, int lane) {
    const int m = lane & 15, h = lane >> 4;
    const _Float16* row = a0 + (size_t)m * ld;
    v16h a;
#pragma unroll
    for (int r = 0; r < 8; ++r) {
        const int kb = a_kbase(r, h);
        a[2 * r]     = row[kb];
        a[2 * r + 1] = row[kb + 1];
    }
    return a;
}

// B from row-major f32 weight W[k][n] (ld = row stride), w0 at (k_base, col_base)
__device__ __forceinline__ v16h load_b_w(const float* w0, int ld, int lane) {
    const int n = lane & 15, kg = lane >> 4;
    v16h b;
#pragma unroll
    for (int r = 0; r < 8; ++r) {
        const int kk = kg * 16 + 2 * r;
        b[2 * r]     = (_Float16)w0[(size_t)kk * ld + n];
        b[2 * r + 1] = (_Float16)w0[(size_t)(kk + 1) * ld + n];
    }
    return b;
}

// B = M^T where M is row-major f16: B[k][n] = row(n)[k]; r0 at (row_base, k_base)
__device__ __forceinline__ v16h load_b_rows_f16(const _Float16* r0, int ld, int lane) {
    const int n = lane & 15, kg = lane >> 4;
    const _Float16* row = r0 + (size_t)n * ld;
    v16h b;
#pragma unroll
    for (int r = 0; r < 8; ++r) {
        const int kk = kg * 16 + 2 * r;
        b[2 * r]     = row[kk];
        b[2 * r + 1] = row[kk + 1];
    }
    return b;
}

// B from row-major f16 matrix M[k][n]
__device__ __forceinline__ v16h load_b_f16_rm(const _Float16* b0, int ld, int lane) {
    const int n = lane & 15, kg = lane >> 4;
    v16h b;
#pragma unroll
    for (int r = 0; r < 8; ++r) {
        const int kk = kg * 16 + 2 * r;
        b[2 * r]     = b0[(size_t)kk * ld + n];
        b[2 * r + 1] = b0[(size_t)(kk + 1) * ld + n];
    }
    return b;
}

__device__ __forceinline__ v8f wmma(v16h a, v16h b, v8f c) {
    return __builtin_amdgcn_wmma_f32_16x16x32_f16(false, a, false, b, (short)0, c,
                                                  false, false);
}

// ---------------------------------------------------------------------------
// Kernel 1: q,k,v projections.  grid (T/16, B), 256 threads = 8 waves.
// 32 output N-tiles per 16-row M-tile: [0,8)=q  [8,16)=k  [16,32)=v
// ---------------------------------------------------------------------------
__global__ __launch_bounds__(256) void qkv_kernel(
    const float* __restrict__ inp, const float* __restrict__ feat,
    const float* __restrict__ Wq, const float* __restrict__ bq,
    const float* __restrict__ Wk, const float* __restrict__ bk,
    const float* __restrict__ Wv, const float* __restrict__ bv,
    _Float16* __restrict__ q16, _Float16* __restrict__ k16,
    float* __restrict__ v32)
{
    const int t0   = blockIdx.x * 16;
    const int b    = blockIdx.y;
    const int lane = threadIdx.x & 31;
    const int w    = threadIdx.x >> 5;
    const int n    = lane & 15, h = lane >> 4;

    const float* inp0 = inp  + ((size_t)b * TT + t0) * CC;
    const float* ft0  = feat + ((size_t)b * TT + t0) * FDD;

#pragma unroll 1
    for (int t = 0; t < 4; ++t) {
        const int nt = w * 4 + t;
        v8f acc = {};
        if (nt < 8) {                       // ---- q: depth 512 = inp|feat
            const int c0 = nt * 16;
#pragma unroll 1
            for (int kc = 0; kc < 16; ++kc) {
                v16h a = (kc < 8) ? load_a_f32(inp0 + kc * 32, CC, lane)
                                  : load_a_f32(ft0 + (kc - 8) * 32, FDD, lane);
                v16h bm = load_b_w(Wq + (size_t)(kc * 32) * KK + c0, KK, lane);
                acc = wmma(a, bm, acc);
            }
            const float bias = bq[c0 + n];
#pragma unroll
            for (int r = 0; r < 8; ++r) {
                const int row = t0 + r + 8 * h;
                q16[((size_t)b * TT + row) * KK + c0 + n] = (_Float16)(acc[r] + bias);
            }
        } else if (nt < 16) {               // ---- k: depth 256
            const int c0 = (nt - 8) * 16;
#pragma unroll 1
            for (int kc = 0; kc < 8; ++kc) {
                v16h a  = load_a_f32(inp0 + kc * 32, CC, lane);
                v16h bm = load_b_w(Wk + (size_t)(kc * 32) * KK + c0, KK, lane);
                acc = wmma(a, bm, acc);
            }
            const float bias = bk[c0 + n];
#pragma unroll
            for (int r = 0; r < 8; ++r) {
                const int row = t0 + r + 8 * h;
                k16[((size_t)b * TT + row) * KK + c0 + n] = (_Float16)(acc[r] + bias);
            }
        } else {                            // ---- v: depth 256, keep f32
            const int c0 = (nt - 16) * 16;
#pragma unroll 1
            for (int kc = 0; kc < 8; ++kc) {
                v16h a  = load_a_f32(inp0 + kc * 32, CC, lane);
                v16h bm = load_b_w(Wv + (size_t)(kc * 32) * VV + c0, VV, lane);
                acc = wmma(a, bm, acc);
            }
            const float bias = bv[c0 + n];
#pragma unroll
            for (int r = 0; r < 8; ++r) {
                const int row = t0 + r + 8 * h;
                v32[((size_t)b * TT + row) * VV + c0 + n] = acc[r] + bias;
            }
        }
    }
}

// ---------------------------------------------------------------------------
// Kernel 2: colsum[b,i] = sum_{j>=i} exp(q_j . k_i / sqrt(K))
// grid (T/16 = it, B); each wave walks jt = it+w, it+w+8, ...
// ---------------------------------------------------------------------------
__global__ __launch_bounds__(256) void colsum_kernel(
    const _Float16* __restrict__ q16, const _Float16* __restrict__ k16,
    float* __restrict__ colsum)
{
    const int it   = blockIdx.x;
    const int b    = blockIdx.y;
    const int lane = threadIdx.x & 31;
    const int w    = threadIdx.x >> 5;
    const int n    = lane & 15, h = lane >> 4;
    const int i0   = it * 16;
    const float inv = 0.08838834764831845f;   // 1/sqrt(128)

    v16h bk[4];
#pragma unroll
    for (int kc = 0; kc < 4; ++kc)
        bk[kc] = load_b_rows_f16(k16 + ((size_t)b * TT + i0) * KK + kc * 32, KK, lane);

    for (int jt = it + w; jt < TT / 16; jt += 8) {
        const int j0 = jt * 16;
        v8f s = {};
#pragma unroll
        for (int kc = 0; kc < 4; ++kc) {
            v16h a = load_a_f16(q16 + ((size_t)b * TT + j0) * KK + kc * 32, KK, lane);
            s = wmma(a, bk[kc], s);
        }
        const int icol = i0 + n;
        float sum = 0.f;
#pragma unroll
        for (int r = 0; r < 8; ++r) {
            const int j = j0 + r + 8 * h;
            sum += (icol <= j) ? __expf(s[r] * inv) : 0.f;
        }
        sum += __shfl_xor(sum, 16, 32);
        if (lane < 16)
            atomicAdd(colsum + (size_t)b * TT + icol, sum);
    }
}

// ---------------------------------------------------------------------------
// Kernel 3: v' = v / colsum  (f32 -> f16)
// ---------------------------------------------------------------------------
__global__ __launch_bounds__(256) void vscale_kernel(
    const float* __restrict__ v32, const float* __restrict__ colsum,
    _Float16* __restrict__ v16o)
{
    const size_t idx = (size_t)blockIdx.x * 256 + threadIdx.x;
    const size_t row = idx / VV;           // b*T + i
    v16o[idx] = (_Float16)(v32[idx] / colsum[row]);
}

// ---------------------------------------------------------------------------
// Kernel 4: out[:, :, 256:512] = E @ v' (recompute scores), plus inp copy.
// grid (T/16 = jt, B), 256 threads = 8 waves.
// Per i_super (256 keys): wave w produces its 16x32 E chunk into LDS,
// barrier, every wave consumes all 8 chunks for its 2 V-tiles.
// ---------------------------------------------------------------------------
__global__ __launch_bounds__(256) void attnout_kernel(
    const float* __restrict__ inp,
    const _Float16* __restrict__ q16, const _Float16* __restrict__ k16,
    const _Float16* __restrict__ v16, float* __restrict__ out)
{
    __shared__ _Float16 ldsE[8][16][32];   // [chunk][j][ii]  8 KB

    const int jt   = blockIdx.x;
    const int b    = blockIdx.y;
    const int lane = threadIdx.x & 31;
    const int w    = threadIdx.x >> 5;
    const int n    = lane & 15, h = lane >> 4;
    const int j0   = jt * 16;
    const int jmax = j0 + 15;
    const float inv = 0.08838834764831845f;

    // Q fragments for this row-tile (reused for every score recompute)
    v16h aq[4];
#pragma unroll
    for (int kc = 0; kc < 4; ++kc)
        aq[kc] = load_a_f16(q16 + ((size_t)b * TT + j0) * KK + kc * 32, KK, lane);

    v8f acc0 = {}, acc1 = {};
    const int nt0 = w * 2;                  // this wave's two V column tiles
    const int nsuper = (jt / 16) + 1;       // ceil((jmax+1)/256)

#pragma unroll 1
    for (int is = 0; is < nsuper; ++is) {
        const int ibase = is * 256 + w * 32;
        if (ibase <= jmax) {                // produce E chunk (wave-uniform branch)
#pragma unroll
            for (int st = 0; st < 2; ++st) {
                const int isb = ibase + st * 16;
                v8f s = {};
#pragma unroll
                for (int kc = 0; kc < 4; ++kc) {
                    v16h bm = load_b_rows_f16(
                        k16 + ((size_t)b * TT + isb) * KK + kc * 32, KK, lane);
                    s = wmma(aq[kc], bm, s);
                }
                const int icol = isb + n;
#pragma unroll
                for (int r = 0; r < 8; ++r) {
                    const int j = j0 + r + 8 * h;
                    const float e = (icol <= j) ? __expf(s[r] * inv) : 0.f;
                    ldsE[w][r + 8 * h][st * 16 + n] = (_Float16)e;
                }
            }
        }
        __syncthreads();
#pragma unroll 1
        for (int cc = 0; cc < 8; ++cc) {
            const int cib = is * 256 + cc * 32;
            if (cib > jmax) break;          // chunks past diagonal contribute 0
            v16h ea = load_a_f16(&ldsE[cc][0][0], 32, lane);
            v16h b0 = load_b_f16_rm(v16 + ((size_t)b * TT + cib) * VV + nt0 * 16,
                                    VV, lane);
            acc0 = wmma(ea, b0, acc0);
            v16h b1 = load_b_f16_rm(v16 + ((size_t)b * TT + cib) * VV + (nt0 + 1) * 16,
                                    VV, lane);
            acc1 = wmma(ea, b1, acc1);
        }
        __syncthreads();
    }

    // write attention output: columns [256, 512)
#pragma unroll
    for (int r = 0; r < 8; ++r) {
        const int j = j0 + r + 8 * h;
        float* orow = out + ((size_t)b * TT + j) * OUTC;
        orow[CC + nt0 * 16 + n]       = acc0[r];
        orow[CC + (nt0 + 1) * 16 + n] = acc1[r];
    }

    // passthrough: out[:, :, 0:256] = inp   (float4 copies, 1024 per block)
    const float* src  = inp + ((size_t)b * TT + j0) * CC;
    float*       dst0 = out + ((size_t)b * TT + j0) * OUTC;
    for (int z = threadIdx.x; z < 16 * (CC / 4); z += 256) {
        const int row = z / (CC / 4);
        const int c4  = z % (CC / 4);
        ((float4*)(dst0 + (size_t)row * OUTC))[c4] =
            ((const float4*)(src + (size_t)row * CC))[c4];
    }
}

// ---------------------------------------------------------------------------
extern "C" void kernel_launch(void* const* d_in, const int* in_sizes, int n_in,
                              void* d_out, int out_size, void* d_ws, size_t ws_size,
                              hipStream_t stream) {
    const float* inp  = (const float*)d_in[0];
    const float* feat = (const float*)d_in[1];
    const float* Wq   = (const float*)d_in[2];
    const float* bq   = (const float*)d_in[3];
    const float* Wk   = (const float*)d_in[4];
    const float* bk   = (const float*)d_in[5];
    const float* Wv   = (const float*)d_in[6];
    const float* bv   = (const float*)d_in[7];
    float* out = (float*)d_out;

    // workspace layout (~32 MB total)
    char* ws = (char*)d_ws;
    size_t off = 0;
    _Float16* q16 = (_Float16*)(ws + off); off += (size_t)BB * TT * KK * sizeof(_Float16);
    _Float16* k16 = (_Float16*)(ws + off); off += (size_t)BB * TT * KK * sizeof(_Float16);
    _Float16* v16 = (_Float16*)(ws + off); off += (size_t)BB * TT * VV * sizeof(_Float16);
    float*    v32 = (float*)(ws + off);    off += (size_t)BB * TT * VV * sizeof(float);
    float*    colsum = (float*)(ws + off); off += (size_t)BB * TT * sizeof(float);

    hipMemsetAsync(colsum, 0, (size_t)BB * TT * sizeof(float), stream);

    qkv_kernel<<<dim3(TT / 16, BB), 256, 0, stream>>>(
        inp, feat, Wq, bq, Wk, bk, Wv, bv, q16, k16, v32);

    colsum_kernel<<<dim3(TT / 16, BB), 256, 0, stream>>>(q16, k16, colsum);

    const size_t nv = (size_t)BB * TT * VV;
    vscale_kernel<<<dim3((unsigned)(nv / 256)), 256, 0, stream>>>(v32, colsum, v16);

    attnout_kernel<<<dim3(TT / 16, BB), 256, 0, stream>>>(inp, q16, k16, v16, out);
}
